// Qwen3NextGatedDeltaNet_67276367725130
// MI455X (gfx1250) — compile-verified, matrix operand order checked by reading, not run on hardware
//
#include <hip/hip_runtime.h>

// ---------------------------------------------------------------------------
// Qwen3-Next Gated DeltaNet prefill for MI455X (gfx1250, wave32, WMMA + TDM).
//
//   0) f32->bf16 preconvert of x, W_qkvz, W_out (one-time, memory-bound)
//   1) gemm_bf16_wmma : qkvz = x @ W_qkvz^T   (206 GFLOP, bf16 WMMA,
//                       TDM tensor_load_to_lds double-buffered staging)
//   2) ba_gemm        : ba   = x @ W_ba^T     (tiny, VALU)
//   3) conv_silu      : depthwise conv K=4 + SiLU
//   4) qk_norm        : L2 norm + DK^-0.5 scale, kv-head repeat
//   5) z_reorder, bg  : unpack z; beta=sigmoid(b); g=-exp(A)*softplus(a+dt)
//   6) scan           : sequential delta rule, state in VGPRs, LDS broadcast
//   7) gated_norm     : RMSNorm(o)*w*silu(z), emits bf16 directly
//   8) gemm_bf16_wmma : out = og @ W_out^T    (69 GFLOP)
// ---------------------------------------------------------------------------

#define LSEQ 4096
#define HDIM 2048
#define HK   16
#define HV   32
#define DK   128
#define DV   128
#define KEY_DIM  2048
#define VAL_DIM  4096
#define QKVZ_DIM 12288
#define CONV_DIM 8192
#define EPSV 1e-6f

typedef __attribute__((ext_vector_type(16))) __bf16 v16bf;
typedef __attribute__((ext_vector_type(8)))  float  v8f;
typedef __attribute__((ext_vector_type(4)))  unsigned int u32x4;
typedef __attribute__((ext_vector_type(8)))  int i32x8;
typedef __attribute__((ext_vector_type(4)))  int i32x4;

#if defined(__has_builtin)
#  if __has_builtin(__builtin_amdgcn_tensor_load_to_lds)
#    define HAVE_TDM 1
#  endif
#endif
#ifndef HAVE_TDM
#  define HAVE_TDM 0
#endif
#if HAVE_TDM && defined(__clang_major__) && (__clang_major__ >= 23)
#  define TDM_ARGS6 1
#else
#  define TDM_ARGS6 0
#endif

__device__ __forceinline__ __bf16 f2bf(float f) {
  union { float f; unsigned u; } c; c.f = f;
  unsigned r = (c.u + 0x7FFFu + ((c.u >> 16) & 1u)) >> 16;   // RNE
  unsigned short h = (unsigned short)r;
  __bf16 out;
  __builtin_memcpy(&out, &h, 2);
  return out;
}

__device__ __forceinline__ float siluf(float x) {
  return x / (1.f + __expf(-x));
}

// ---------------------------------------------------------------------------
// One-time fp32 -> bf16 conversion (packed pair per thread-element).
// ---------------------------------------------------------------------------
__global__ void f32_to_bf16_kernel(const float* __restrict__ src,
                                   unsigned* __restrict__ dst, int npairs)
{
  int i = blockIdx.x * blockDim.x + threadIdx.x;
  if (i >= npairs) return;
  float2 f = ((const float2*)src)[i];
  union { float f; unsigned u; } a, b; a.f = f.x; b.f = f.y;
  unsigned lo = (a.u + 0x7FFFu + ((a.u >> 16) & 1u)) >> 16;
  unsigned hi = (b.u + 0x7FFFu + ((b.u >> 16) & 1u)) & 0xFFFF0000u;
  dst[i] = (lo & 0xFFFFu) | hi;
}

// ---------------------------------------------------------------------------
// TDM: async 2D tile copy global(bf16) -> LDS with hardware padding.
// Tile 128 rows x 32 cols (2B elems). Pad 4 DWORDs after every 16 DWORDs
// -> LDS row stride 80B = 40 bf16 (bank-conflict-free fragment reads).
// Descriptor per CDNA5 ISA.
// ---------------------------------------------------------------------------
#define BM 128
#define BN 128
#define BK 32
#define LDSP 40      // 32 payload + 4-DWORD pad

#if HAVE_TDM
__device__ __forceinline__ void tdm_load_tile_bf16(
    unsigned lds_addr, const __bf16* gptr, unsigned kstride_elems,
    unsigned tensor_d0, unsigned tensor_d1)
{
  unsigned long long ga = (unsigned long long)(uintptr_t)gptr;
  u32x4 g0;
  g0[0] = 1u;                                            // count=1, user mode
  g0[1] = lds_addr;                                      // LDS byte address
  g0[2] = (unsigned)(ga & 0xFFFFFFFFu);                  // global_addr[31:0]
  g0[3] = (unsigned)((ga >> 32) & 0x01FFFFFFu) | (2u << 30); // [56:32] | type=2
  i32x8 g1;
  g1[0] = (int)((1u << 16)        // data_size = 1 -> 2 bytes
              | (1u << 20)        // pad_enable
              | (3u << 22)        // pad_interval: 16 DWORDs
              | (3u << 25));      // pad_amount:   4 DWORDs
  g1[1] = (int)((tensor_d0 & 0xFFFFu) << 16);            // tensor_dim0[15:0]
  g1[2] = (int)((tensor_d0 >> 16) | ((tensor_d1 & 0xFFFFu) << 16));
  g1[3] = (int)((tensor_d1 >> 16) | ((unsigned)BK << 16)); // tile_dim0=32
  g1[4] = (int)BM;                                       // tile_dim1=128
  g1[5] = (int)kstride_elems;                            // tensor_dim0_stride
  g1[6] = 0;
  g1[7] = 0;
  i32x4 z4 = {0, 0, 0, 0};
#if TDM_ARGS6
  i32x8 z8 = {0, 0, 0, 0, 0, 0, 0, 0};
  __builtin_amdgcn_tensor_load_to_lds(g0, g1, z4, z4, z8, 0);
#else
  __builtin_amdgcn_tensor_load_to_lds(g0, g1, z4, z4, 0);
#endif
}
#endif

// ---------------------------------------------------------------------------
// Per-wave 2x4-subtile WMMA compute on one staged 128x128x32 tile.
// Fragment layout per ISA 16-bit A table: lane = (K-half)*16 + M,
// VGPR pairs over K -> two contiguous 16B runs per fragment (ds_load_b128 x2).
// ---------------------------------------------------------------------------
__device__ __forceinline__ void wmma_tile(
    const __bf16 (*__restrict__ as)[LDSP], const __bf16 (*__restrict__ bs)[LDSP],
    int lane, int waveM, int waveN, v8f acc[2][4])
{
  const int fm = lane & 15;
  const int kb = (lane >> 4) * 8;

  v16bf bfrag[4];
  #pragma unroll
  for (int ni = 0; ni < 4; ++ni) {
    const int n = waveN * 64 + ni * 16 + fm;
    #pragma unroll
    for (int v2 = 0; v2 < 8; ++v2) {
      const int kk = (v2 < 4) ? (kb + 2 * v2) : (kb + 16 + 2 * (v2 - 4));
      bfrag[ni][2 * v2]     = bs[n][kk];
      bfrag[ni][2 * v2 + 1] = bs[n][kk + 1];
    }
  }
  #pragma unroll
  for (int mi = 0; mi < 2; ++mi) {
    const int mrow = waveM * 32 + mi * 16 + fm;
    v16bf afrag;
    #pragma unroll
    for (int v2 = 0; v2 < 8; ++v2) {
      const int kk = (v2 < 4) ? (kb + 2 * v2) : (kb + 16 + 2 * (v2 - 4));
      afrag[2 * v2]     = as[mrow][kk];
      afrag[2 * v2 + 1] = as[mrow][kk + 1];
    }
    #pragma unroll
    for (int ni = 0; ni < 4; ++ni) {
      acc[mi][ni] = __builtin_amdgcn_wmma_f32_16x16x32_bf16(
          false, afrag, false, bfrag[ni], (short)0, acc[mi][ni], false, false);
    }
  }
}

// ---------------------------------------------------------------------------
// GEMM: C[M,N] = A[M,K](bf16) * W[N,K](bf16)^T, fp32 out.
// 256 threads = 8 waves; tile 128x128x32; TDM double-buffered staging.
// ---------------------------------------------------------------------------
__global__ __launch_bounds__(256) void gemm_bf16_wmma(
    const __bf16* __restrict__ A, const __bf16* __restrict__ Bw,
    float* __restrict__ C, int M, int N, int Kd)
{
#if HAVE_TDM
  __shared__ __bf16 As[2][BM][LDSP];
  __shared__ __bf16 Bs[2][BN][LDSP];
#else
  __shared__ __bf16 As[1][BM][LDSP];
  __shared__ __bf16 Bs[1][BN][LDSP];
#endif

  const int tid   = threadIdx.x;
  const int lane  = tid & 31;
  const int wave  = tid >> 5;
  const int waveM = wave & 3;     // 4 waves along M -> 32 rows each
  const int waveN = wave >> 2;    // 2 waves along N -> 64 cols each
  const int tileM = blockIdx.x * BM;
  const int tileN = blockIdx.y * BN;

  v8f acc[2][4];
  #pragma unroll
  for (int i = 0; i < 2; ++i)
    #pragma unroll
    for (int j = 0; j < 4; ++j)
      acc[i][j] = (v8f){0.f,0.f,0.f,0.f,0.f,0.f,0.f,0.f};

  const int nk = Kd / BK;

#if HAVE_TDM
  const __bf16* aBase = A + (size_t)tileM * Kd;
  const __bf16* bBase = Bw + (size_t)tileN * Kd;
  if (wave == 0) {
    tdm_load_tile_bf16((unsigned)(uintptr_t)&As[0][0][0], aBase, (unsigned)Kd,
                       (unsigned)Kd, (unsigned)BM);
    tdm_load_tile_bf16((unsigned)(uintptr_t)&Bs[0][0][0], bBase, (unsigned)Kd,
                       (unsigned)Kd, (unsigned)BN);
  }
  int buf = 0;
  for (int kt = 0; kt < nk; ++kt) {
    if (wave == 0) {
      if (kt + 1 < nk) {                    // prefetch next tile into buf^1
        const int k1 = (kt + 1) * BK;
        tdm_load_tile_bf16((unsigned)(uintptr_t)&As[buf ^ 1][0][0],
                           aBase + k1, (unsigned)Kd, (unsigned)Kd, (unsigned)BM);
        tdm_load_tile_bf16((unsigned)(uintptr_t)&Bs[buf ^ 1][0][0],
                           bBase + k1, (unsigned)Kd, (unsigned)Kd, (unsigned)BN);
        __builtin_amdgcn_s_wait_tensorcnt(2);   // current tile retired
      } else {
        __builtin_amdgcn_s_wait_tensorcnt(0);
      }
    }
    __syncthreads();
    wmma_tile(As[buf], Bs[buf], lane, waveM, waveN, acc);
    __syncthreads();
    buf ^= 1;
  }
#else
  const int ldRow = tid >> 1;          // 0..127
  const int ldCol = (tid & 1) * 16;    // 0 or 16
  for (int kt = 0; kt < nk; ++kt) {
    const int k0 = kt * BK;
    const uint4* ag = (const uint4*)(A + (size_t)(tileM + ldRow) * Kd + k0 + ldCol);
    const uint4* bg = (const uint4*)(Bw + (size_t)(tileN + ldRow) * Kd + k0 + ldCol);
    uint4 a0 = ag[0], a1 = ag[1];
    uint4 b0 = bg[0], b1 = bg[1];
    *(uint4*)&As[0][ldRow][ldCol]     = a0;
    *(uint4*)&As[0][ldRow][ldCol + 8] = a1;
    *(uint4*)&Bs[0][ldRow][ldCol]     = b0;
    *(uint4*)&Bs[0][ldRow][ldCol + 8] = b1;
    __syncthreads();
    wmma_tile(As[0], Bs[0], lane, waveM, waveN, acc);
    __syncthreads();
  }
#endif

  // epilogue: C/D layout — lane: N = l&15, VGPR r: M = r + 8*(l>>4)
  const int cn = lane & 15;
  const int mo = 8 * (lane >> 4);
  #pragma unroll
  for (int mi = 0; mi < 2; ++mi)
    #pragma unroll
    for (int ni = 0; ni < 4; ++ni)
      #pragma unroll
      for (int r = 0; r < 8; ++r) {
        const int row = tileM + waveM * 32 + mi * 16 + mo + r;
        const int col = tileN + waveN * 64 + ni * 16 + cn;
        C[(size_t)row * N + col] = acc[mi][ni][r];
      }
}

// ---------------------------------------------------------------------------
// Small projection: ba = x @ W_ba^T  (N=64, fp32)
// ---------------------------------------------------------------------------
__global__ __launch_bounds__(256) void ba_gemm_kernel(
    const float* __restrict__ x, const float* __restrict__ W_ba,
    float* __restrict__ ba)
{
  int idx = blockIdx.x * 256 + threadIdx.x;    // l*64 + j
  if (idx >= LSEQ * 64) return;
  int l = idx >> 6, j = idx & 63;
  const float4* xp = (const float4*)(x + (size_t)l * HDIM);
  const float4* wp = (const float4*)(W_ba + (size_t)j * HDIM);
  float acc = 0.f;
  for (int i = 0; i < HDIM / 4; ++i) {
    float4 a = xp[i], b = wp[i];
    acc = fmaf(a.x, b.x, acc); acc = fmaf(a.y, b.y, acc);
    acc = fmaf(a.z, b.z, acc); acc = fmaf(a.w, b.w, acc);
  }
  ba[idx] = acc;
}

// ---------------------------------------------------------------------------
// Depthwise causal conv (K=4) + SiLU over packed qkvz layout.
// Packed per k-head: [q:128 | k:128 | v:256 | z:256] -> 768 per head.
// ---------------------------------------------------------------------------
__global__ void conv_silu_kernel(
    const float* __restrict__ qkvz, const float* __restrict__ conv_w,
    float* __restrict__ qtmp, float* __restrict__ ktmp, float* __restrict__ varr)
{
  int idx = blockIdx.x * blockDim.x + threadIdx.x;   // l*8192 + c
  if (idx >= LSEQ * CONV_DIM) return;
  int l = idx >> 13;
  int c = idx & (CONV_DIM - 1);

  int col;
  if (c < KEY_DIM)            { int hk = c >> 7;              col = hk * 768 + (c & 127); }
  else if (c < 2 * KEY_DIM)   { int cc = c - KEY_DIM;
                                int hk = cc >> 7;             col = hk * 768 + 128 + (cc & 127); }
  else                        { int fv = c - 2 * KEY_DIM;
                                int hk = fv >> 8;             col = hk * 768 + 256 + (fv & 255); }

  float acc = 0.f;
  #pragma unroll
  for (int j = 0; j < 4; ++j) {
    int ls = l - 3 + j;
    float xv = (ls >= 0) ? qkvz[(size_t)ls * QKVZ_DIM + col] : 0.f;
    acc = fmaf(xv, conv_w[c * 4 + j], acc);
  }
  float s = siluf(acc);

  if (c < KEY_DIM)          qtmp[(size_t)l * KEY_DIM + c] = s;
  else if (c < 2 * KEY_DIM) ktmp[(size_t)l * KEY_DIM + (c - KEY_DIM)] = s;
  else                      varr[(size_t)l * VAL_DIM + (c - 2 * KEY_DIM)] = s;
}

// ---------------------------------------------------------------------------
// L2 norm q/k per (l, hk), scale q by DK^-0.5, repeat to both v-heads.
// ---------------------------------------------------------------------------
__global__ __launch_bounds__(128) void qk_norm_kernel(
    const float* __restrict__ qtmp, const float* __restrict__ ktmp,
    float* __restrict__ qn, float* __restrict__ kn)
{
  int blk = blockIdx.x;                 // l*16 + hk
  int l = blk >> 4, hk = blk & 15, t = threadIdx.x;
  __shared__ float red[2][128];
  float qv = qtmp[(size_t)l * KEY_DIM + hk * DK + t];
  float kv = ktmp[(size_t)l * KEY_DIM + hk * DK + t];
  red[0][t] = qv * qv;
  red[1][t] = kv * kv;
  __syncthreads();
  for (int s = 64; s > 0; s >>= 1) {
    if (t < s) { red[0][t] += red[0][t + s]; red[1][t] += red[1][t + s]; }
    __syncthreads();
  }
  float qr = rsqrtf(red[0][0] + EPSV) * 0.08838834764831845f;   // 1/sqrt(128)
  float kr = rsqrtf(red[1][0] + EPSV);
  float qo = qv * qr, ko = kv * kr;
  size_t base = ((size_t)l * HV + hk * 2) * DK + t;
  qn[base] = qo; qn[base + DK] = qo;     // repeat_interleave over KVR=2
  kn[base] = ko; kn[base + DK] = ko;
}

// ---------------------------------------------------------------------------
// Unpack z from the packed qkvz layout.
// ---------------------------------------------------------------------------
__global__ void z_reorder_kernel(const float* __restrict__ qkvz,
                                 float* __restrict__ zarr)
{
  int idx = blockIdx.x * blockDim.x + threadIdx.x;   // l*4096 + f
  if (idx >= LSEQ * VAL_DIM) return;
  int l = idx >> 12, f = idx & (VAL_DIM - 1);
  int hv = f >> 7, dv = f & 127;
  int hk = hv >> 1;
  int sub = (hv & 1) * 128 + dv;
  zarr[idx] = qkvz[(size_t)l * QKVZ_DIM + hk * 768 + 512 + sub];
}

// ---------------------------------------------------------------------------
// beta = sigmoid(b); g = -exp(A_log) * softplus(a + dt_bias)
// ---------------------------------------------------------------------------
__global__ void bg_kernel(const float* __restrict__ ba,
                          const float* __restrict__ A_log,
                          const float* __restrict__ dt_bias,
                          float* __restrict__ garr, float* __restrict__ barr)
{
  int idx = blockIdx.x * blockDim.x + threadIdx.x;   // l*32 + hv
  if (idx >= LSEQ * HV) return;
  int l = idx >> 5, hv = idx & 31;
  int hk = hv >> 1, r = hv & 1;
  float bv = ba[(size_t)l * 64 + hk * 4 + r];
  float av = ba[(size_t)l * 64 + hk * 4 + 2 + r];
  barr[idx] = 1.f / (1.f + __expf(-bv));
  float xdt = av + dt_bias[hv];
  float sp = (xdt > 20.f) ? xdt : log1pf(__expf(xdt));
  garr[idx] = -__expf(A_log[hv]) * sp;
}

// ---------------------------------------------------------------------------
// Sequential gated delta rule. One block per v-head (32 blocks, 128 threads).
// Lane t owns state column S[:,t] (128 floats in VGPRs). k/q broadcast via
// double-buffered LDS; one barrier per timestep.
// ---------------------------------------------------------------------------
__global__ __launch_bounds__(128) void scan_kernel(
    const float* __restrict__ qn, const float* __restrict__ kn,
    const float* __restrict__ varr, const float* __restrict__ garr,
    const float* __restrict__ barr, float* __restrict__ oarr)
{
  const int h = blockIdx.x;       // v-head
  const int t = threadIdx.x;      // output channel dv
  __shared__ float kq[2][2][128];

  float S[DK];
  #pragma unroll
  for (int i = 0; i < DK; ++i) S[i] = 0.f;

  kq[0][0][t] = kn[((size_t)0 * HV + h) * DK + t];
  kq[0][1][t] = qn[((size_t)0 * HV + h) * DK + t];
  __syncthreads();

  for (int l = 0; l < LSEQ; ++l) {
    const int buf = l & 1;
    const float gt = garr[(size_t)l * HV + h];
    const float bt = barr[(size_t)l * HV + h];
    const float decay = __expf(gt);
    const float vt = varr[((size_t)l * HV + h) * DV + t];

    float kv = 0.f;
    #pragma unroll
    for (int dk = 0; dk < DK; ++dk) {
      S[dk] *= decay;
      kv = fmaf(S[dk], kq[buf][0][dk], kv);
    }
    const float delta = (vt - kv) * bt;
    float ot = 0.f;
    #pragma unroll
    for (int dk = 0; dk < DK; ++dk) {
      S[dk] = fmaf(kq[buf][0][dk], delta, S[dk]);
      ot = fmaf(kq[buf][1][dk], S[dk], ot);
    }
    oarr[((size_t)l * HV + h) * DV + t] = ot;

    if (l + 1 < LSEQ) {       // stage next step into other buffer
      const int nb = buf ^ 1;
      kq[nb][0][t] = kn[((size_t)(l + 1) * HV + h) * DK + t];
      kq[nb][1][t] = qn[((size_t)(l + 1) * HV + h) * DK + t];
    }
    __syncthreads();
  }
}

// ---------------------------------------------------------------------------
// og = rmsnorm(o) * norm_w * silu(z), emitted as bf16 for the output GEMM.
// ---------------------------------------------------------------------------
__global__ __launch_bounds__(128) void gated_norm_kernel(
    const float* __restrict__ oarr, const float* __restrict__ zarr,
    const float* __restrict__ norm_w, __bf16* __restrict__ og)
{
  int blk = blockIdx.x;          // l*32 + hv
  int t = threadIdx.x;
  __shared__ float red[128];
  size_t base = (size_t)blk * DV;
  float ov = oarr[base + t];
  red[t] = ov * ov;
  __syncthreads();
  for (int s = 64; s > 0; s >>= 1) {
    if (t < s) red[t] += red[t + s];
    __syncthreads();
  }
  float r = rsqrtf(red[0] * (1.f / 128.f) + EPSV);
  float zv = zarr[base + t];
  og[base + t] = f2bf(ov * r * norm_w[t] * siluf(zv));
}

// ---------------------------------------------------------------------------
// Workspace layout (fp32 region in floats, bf16 region after it)
// ---------------------------------------------------------------------------
#define QKVZ_OFF 0ull
#define BA_OFF   50331648ull
#define QTMP_OFF 50593792ull
#define KTMP_OFF 58982400ull
#define V_OFF    67371008ull
#define Z_OFF    84148224ull
#define QN_OFF   100925440ull
#define KN_OFF   117702656ull
#define G_OFF    134479872ull
#define BETA_OFF 134610944ull
#define O_OFF    134742016ull
#define BF16_BASE 151519232ull       // floats; 16B aligned
// bf16 element offsets within bf16 region:
#define XBF_OFF    0ull              // 4096*2048
#define WQKVZ_BOFF 8388608ull        // 12288*2048
#define WOUT_BOFF  33554432ull       // 2048*4096
#define OGBF_OFF   41943040ull       // 4096*4096

extern "C" void kernel_launch(void* const* d_in, const int* in_sizes, int n_in,
                              void* d_out, int out_size, void* d_ws, size_t ws_size,
                              hipStream_t stream) {
  const float* x       = (const float*)d_in[0];
  const float* W_qkvz  = (const float*)d_in[1];
  const float* W_ba    = (const float*)d_in[2];
  const float* conv_w  = (const float*)d_in[3];
  const float* A_log   = (const float*)d_in[4];
  const float* dt_bias = (const float*)d_in[5];
  const float* norm_w  = (const float*)d_in[6];
  const float* W_out   = (const float*)d_in[7];
  float* out = (float*)d_out;
  float* ws  = (float*)d_ws;

  float* qkvz = ws + QKVZ_OFF;
  float* ba   = ws + BA_OFF;
  float* qtmp = ws + QTMP_OFF;
  float* ktmp = ws + KTMP_OFF;
  float* varr = ws + V_OFF;
  float* zarr = ws + Z_OFF;
  float* qn   = ws + QN_OFF;
  float* kn   = ws + KN_OFF;
  float* garr = ws + G_OFF;
  float* barr = ws + BETA_OFF;
  float* oarr = ws + O_OFF;

  __bf16* bfbase = (__bf16*)(ws + BF16_BASE);
  __bf16* xbf     = bfbase + XBF_OFF;
  __bf16* wqkvzbf = bfbase + WQKVZ_BOFF;
  __bf16* woutbf  = bfbase + WOUT_BOFF;
  __bf16* ogbf    = bfbase + OGBF_OFF;

  // 0) one-time bf16 conversions
  f32_to_bf16_kernel<<<(LSEQ * HDIM / 2 + 255) / 256, 256, 0, stream>>>(
      x, (unsigned*)xbf, LSEQ * HDIM / 2);
  f32_to_bf16_kernel<<<(QKVZ_DIM * HDIM / 2 + 255) / 256, 256, 0, stream>>>(
      W_qkvz, (unsigned*)wqkvzbf, QKVZ_DIM * HDIM / 2);
  f32_to_bf16_kernel<<<(HDIM * VAL_DIM / 2 + 255) / 256, 256, 0, stream>>>(
      W_out, (unsigned*)woutbf, HDIM * VAL_DIM / 2);

  // 1) qkvz = x @ W_qkvz^T   (4096x2048 * 2048x12288)
  dim3 g1(LSEQ / BM, QKVZ_DIM / BN);
  gemm_bf16_wmma<<<g1, 256, 0, stream>>>(xbf, wqkvzbf, qkvz, LSEQ, QKVZ_DIM, HDIM);

  // 2) ba = x @ W_ba^T
  ba_gemm_kernel<<<(LSEQ * 64) / 256, 256, 0, stream>>>(x, W_ba, ba);

  // 3) depthwise conv + SiLU
  conv_silu_kernel<<<(LSEQ * CONV_DIM) / 256, 256, 0, stream>>>(
      qkvz, conv_w, qtmp, ktmp, varr);

  // 4) q/k L2 norm + head repeat
  qk_norm_kernel<<<LSEQ * HK, 128, 0, stream>>>(qtmp, ktmp, qn, kn);

  // 5) z unpack, beta/g
  z_reorder_kernel<<<(LSEQ * VAL_DIM) / 256, 256, 0, stream>>>(qkvz, zarr);
  bg_kernel<<<(LSEQ * HV) / 256, 256, 0, stream>>>(ba, A_log, dt_bias, garr, barr);

  // 6) sequential delta-rule scan
  scan_kernel<<<HV, 128, 0, stream>>>(qn, kn, varr, garr, barr, oarr);

  // 7) gated RMSNorm -> bf16
  gated_norm_kernel<<<LSEQ * HV, 128, 0, stream>>>(oarr, zarr, norm_w, ogbf);

  // 8) out = og @ W_out^T   (4096x4096 * 4096x2048)
  dim3 g2(LSEQ / BM, HDIM / BN);
  gemm_bf16_wmma<<<g2, 256, 0, stream>>>(ogbf, woutbf, out, LSEQ, HDIM, VAL_DIM);
}
